// EpisodicMemoryModule_7876970021233
// MI455X (gfx1250) — matrix operand order repaired and checked by reading, not compiled, not run on hardware
//
#include <hip/hip_runtime.h>
#include <cstdint>

// ---------------------------------------------------------------------------
// Problem constants (from the reference)
// ---------------------------------------------------------------------------
#define MMEM 65536
#define KD 512
#define VD 512
#define BS 2
#define NQ 1024      // BS*NR
#define NKROWS 128   // BS*NW
#define CHUNKSZ 8
#define NCHUNK 16    // NKROWS/CHUNKSZ
#define TKW 5
#define TKR 10
#define G4 2048      // 4*VD

typedef float v2f __attribute__((ext_vector_type(2)));
typedef float v8f __attribute__((ext_vector_type(8)));
typedef __bf16 v16bf __attribute__((ext_vector_type(16)));
typedef unsigned int v4u __attribute__((ext_vector_type(4)));
typedef int v4i __attribute__((ext_vector_type(4)));
typedef int v8i __attribute__((ext_vector_type(8)));
typedef unsigned short u16;

// ---------------------------------------------------------------------------
// Helpers
// ---------------------------------------------------------------------------
__device__ __forceinline__ u16 f2bf(float x) {
  unsigned u = __float_as_uint(x);
  unsigned r = (u + 0x7FFFu + ((u >> 16) & 1u)) >> 16; // RNE
  return (u16)r;
}
__device__ __forceinline__ float bf2f(u16 h) { return __uint_as_float(((unsigned)h) << 16); }
__device__ __forceinline__ float sigmf(float x) { return 1.0f / (1.0f + __expf(-x)); }

// register-resident top-K insertion (tie-break: lower index first, like jax top_k)
template <int N>
__device__ __forceinline__ void tk_ins(float (&tv)[N], int (&ti)[N], float cv, int ci) {
#pragma unroll
  for (int p = 0; p < N; ++p) {
    bool bt = (cv > tv[p]) || (cv == tv[p] && ci < ti[p]);
    float ov = tv[p]; int oi = ti[p];
    if (bt) { tv[p] = cv; ti[p] = ci; cv = ov; ci = oi; }
  }
}

// ---------------------------------------------------------------------------
// Split-bf16 (hi+lo) multi-tile accumulate:
//   acc[t] += sum_k A[arow+m][k] * B[brow0+t*bstep+n][k]
// A fragments loaded once per K-step and shared by NT tiles; NT independent
// accumulator chains keep the XDL pipeline busy. 3 bf16 WMMAs ~ f32 accuracy.
// ---------------------------------------------------------------------------
template <int NT>
__device__ __forceinline__ void wmma_bf16x3_multi(
    const u16* __restrict__ Ahi, const u16* __restrict__ Alo, int lda, int arow,
    const u16* __restrict__ Bhi, const u16* __restrict__ Blo, int ldb, int brow0, int bstep,
    int Kr, int lane, v8f (&acc)[NT]) {
  const int half = lane >> 4, sub = lane & 15;
  const u16* pah = Ahi + (size_t)(arow + sub) * lda;
  const u16* pal = Alo + (size_t)(arow + sub) * lda;
  const u16* pbh[NT]; const u16* pbl[NT];
#pragma unroll
  for (int t = 0; t < NT; ++t) {
    pbh[t] = Bhi + (size_t)(brow0 + t * bstep + sub) * ldb;
    pbl[t] = Blo + (size_t)(brow0 + t * bstep + sub) * ldb;
  }
  for (int k = 0; k < Kr; k += 32) {
    union Frag { v4u q[2]; v16bf v; } ah, al;
    // A frag: lanes 0-15 K=[0..7]+[16..23], lanes 16-31 K=[8..15]+[24..31]
    ah.q[0] = *(const v4u*)(pah + k + half * 8);
    ah.q[1] = *(const v4u*)(pah + k + 16 + half * 8);
    al.q[0] = *(const v4u*)(pal + k + half * 8);
    al.q[1] = *(const v4u*)(pal + k + 16 + half * 8);
    __builtin_prefetch(pah + k + 128, 0, 1);
#pragma unroll
    for (int t = 0; t < NT; ++t) {
      union Frag bh, bl;
      // B frag: lanes 0-15 K=[0..7]+[16..23], lanes 16-31 K=[8..15]+[24..31]
      bh.q[0] = *(const v4u*)(pbh[t] + k + half * 8);
      bh.q[1] = *(const v4u*)(pbh[t] + k + 16 + half * 8);
      bl.q[0] = *(const v4u*)(pbl[t] + k + half * 8);
      bl.q[1] = *(const v4u*)(pbl[t] + k + 16 + half * 8);
      acc[t] = __builtin_amdgcn_wmma_f32_16x16x32_bf16(false, ah.v, false, bh.v, (short)0, acc[t], false, false);
      acc[t] = __builtin_amdgcn_wmma_f32_16x16x32_bf16(false, ah.v, false, bl.v, (short)0, acc[t], false, false);
      acc[t] = __builtin_amdgcn_wmma_f32_16x16x32_bf16(false, al.v, false, bh.v, (short)0, acc[t], false, false);
    }
  }
}

// ---------------------------------------------------------------------------
// Exact f32 GEMM via V_WMMA_F32_16X16X4_F32:  C[m,n] = scale*sum_k A[m,k]*W[n,k] + b0[n]+b1[n]
// One wave per 16x16 tile.
// ---------------------------------------------------------------------------
__global__ __launch_bounds__(32) void gemm_f32_wmma(
    const float* __restrict__ A, int lda,
    const float* __restrict__ W, int ldw,
    float* __restrict__ C, int ldc,
    int Kr, float scale,
    const float* __restrict__ bias0, const float* __restrict__ bias1) {
  const int l = threadIdx.x;
  const int half = l >> 4, sub = l & 15;
  const int tileN = blockIdx.x * 16, tileM = blockIdx.y * 16;
  const float* ar = A + (size_t)(tileM + sub) * lda + half * 2;
  const float* wr = W + (size_t)(tileN + sub) * ldw + half * 2;
  v8f acc = {};
  for (int k = 0; k < Kr; k += 4) {
    v2f a = *(const v2f*)(ar + k);
    v2f b = *(const v2f*)(wr + k);
    acc = __builtin_amdgcn_wmma_f32_16x16x4_f32(false, a, false, b, (short)0, acc, false, false);
  }
  const int col = tileN + sub;
  float bv = 0.f;
  if (bias0) bv += bias0[col];
  if (bias1) bv += bias1[col];
#pragma unroll
  for (int i = 0; i < 8; ++i) {
    const int row = tileM + i + half * 8;
    C[(size_t)row * ldc + col] = acc[i] * scale + bv;
  }
}

// ---------------------------------------------------------------------------
// Row-wise f32 -> bf16 hi/lo split + L2 norm (wave per row)
// ---------------------------------------------------------------------------
__global__ __launch_bounds__(256) void prep_rows(
    const float* __restrict__ src, u16* __restrict__ hi, u16* __restrict__ lo,
    float* __restrict__ norms, int ncols) {
  const int row = blockIdx.x * 8 + (threadIdx.x >> 5);
  const int lane = threadIdx.x & 31;
  const float* s = src + (size_t)row * ncols;
  float ss = 0.f;
  for (int c = lane; c < ncols; c += 32) {
    float x = s[c];
    u16 h = f2bf(x);
    hi[(size_t)row * ncols + c] = h;
    lo[(size_t)row * ncols + c] = f2bf(x - bf2f(h));
    ss += x * x;
  }
  for (int o = 16; o > 0; o >>= 1) ss += __shfl_xor(ss, o, 32);
  if (lane == 0) norms[row] = sqrtf(ss);
}

// ---------------------------------------------------------------------------
// Hebbian: per-chunk key prep (8 rows + 8 zero pad rows), hi/lo + norms
// ---------------------------------------------------------------------------
__global__ __launch_bounds__(256) void chunk_prep(
    const float* __restrict__ kp, int chunk,
    u16* __restrict__ kchh, u16* __restrict__ kchl, float* __restrict__ kn) {
  const int w = threadIdx.x >> 5, lane = threadIdx.x & 31;
  const float* src = kp + (size_t)(chunk * CHUNKSZ + w) * KD;
  float ss = 0.f;
  for (int c = lane; c < KD; c += 32) {
    float x = src[c];
    u16 h = f2bf(x);
    kchh[w * KD + c] = h;
    kchl[w * KD + c] = f2bf(x - bf2f(h));
    kchh[(w + 8) * KD + c] = 0;
    kchl[(w + 8) * KD + c] = 0;
    ss += x * x;
  }
  for (int o = 16; o > 0; o >>= 1) ss += __shfl_xor(ss, o, 32);
  if (lane == 0) kn[w] = sqrtf(ss);
}

// ---------------------------------------------------------------------------
// Hebbian: cosine similarity of the 8 chunk keys vs all 65536 memory keys (WMMA)
// ---------------------------------------------------------------------------
__global__ __launch_bounds__(256) void hebb_sim(
    const u16* __restrict__ kchh, const u16* __restrict__ kchl, const float* __restrict__ kn,
    const u16* __restrict__ mkh, const u16* __restrict__ mkl, const float* __restrict__ normk,
    float* __restrict__ sim) {
  const int lane = threadIdx.x & 31, w = threadIdx.x >> 5;
  const int colbase = blockIdx.x * 2048 + w * 16;
  for (int g = 0; g < 4; ++g) {
    v8f acc[4] = {};
    wmma_bf16x3_multi<4>(kchh, kchl, KD, 0, mkh, mkl, KD, colbase + g * 512, 128, KD, lane, acc);
    if (lane < 16) {  // lanes 0-15 hold rows 0..7 = the real chunk keys
#pragma unroll
      for (int u = 0; u < 4; ++u) {
        const int c = colbase + g * 512 + u * 128 + lane;
        const float nm = fmaxf(normk[c], 1e-12f);
#pragma unroll
        for (int i = 0; i < 8; ++i) {
          const float nk = fmaxf(kn[i], 1e-12f);
          sim[(size_t)i * MMEM + c] = acc[u][i] / (nk * nm);
        }
      }
    }
  }
}

// ---------------------------------------------------------------------------
// Hebbian: top-5 per chunk row + alpha = lr*sigmoid(sim)
// ---------------------------------------------------------------------------
__global__ __launch_bounds__(256) void hebb_top5(
    const float* __restrict__ sim, const float* __restrict__ lr,
    int* __restrict__ cidx, float* __restrict__ calpha) {
  __shared__ float sv[256 * TKW];
  __shared__ int si[256 * TKW];
  const int i = blockIdx.x;
  float tv[TKW]; int ti[TKW];
#pragma unroll
  for (int s = 0; s < TKW; ++s) { tv[s] = -3.4e38f; ti[s] = 0x7fffffff; }
  for (int j = threadIdx.x; j < MMEM; j += 256) tk_ins<TKW>(tv, ti, sim[(size_t)i * MMEM + j], j);
#pragma unroll
  for (int s = 0; s < TKW; ++s) { sv[threadIdx.x * TKW + s] = tv[s]; si[threadIdx.x * TKW + s] = ti[s]; }
  __syncthreads();
  if (threadIdx.x == 0) {
    float mv_[TKW]; int mi[TKW];
#pragma unroll
    for (int s = 0; s < TKW; ++s) { mv_[s] = -3.4e38f; mi[s] = 0x7fffffff; }
    for (int c = 0; c < 256 * TKW; ++c) tk_ins<TKW>(mv_, mi, sv[c], si[c]);
    const float L = lr[0];
#pragma unroll
    for (int s = 0; s < TKW; ++s) { cidx[i * TKW + s] = mi[s]; calpha[i * TKW + s] = L * sigmf(mv_[s]); }
  }
}

// ---------------------------------------------------------------------------
// Hebbian: sequential 40 row blends, then refresh bf16 mirrors + norms of
// the touched rows (so the next chunk / the read phase see updated keys).
// ---------------------------------------------------------------------------
__global__ __launch_bounds__(256) void hebb_update(
    float* __restrict__ mk, float* __restrict__ mvv,
    u16* __restrict__ mkh, u16* __restrict__ mkl, float* __restrict__ normk,
    const float* __restrict__ kp, const float* __restrict__ vp,
    const int* __restrict__ cidx, const float* __restrict__ calpha, int chunk) {
  const int tid = threadIdx.x;
  for (int t = 0; t < CHUNKSZ * TKW; ++t) {
    const int i = t / TKW;
    const int idx = cidx[t];
    const float a = calpha[t];
    const float* krow = kp + (size_t)(chunk * CHUNKSZ + i) * KD;
    const float* vrow = vp + (size_t)(chunk * CHUNKSZ + i) * VD;
    float* mkr = mk + (size_t)idx * KD;
    float* mvr = mvv + (size_t)idx * VD;
    for (int c = tid; c < KD; c += 256) {
      mkr[c] = (1.f - a) * mkr[c] + a * krow[c];
      mvr[c] = (1.f - a) * mvr[c] + a * vrow[c];
    }
    __syncthreads();
  }
  __shared__ float red[256];
  for (int t = 0; t < CHUNKSZ * TKW; ++t) {
    const int idx = cidx[t];
    float ss = 0.f;
    for (int c = tid; c < KD; c += 256) {
      float x = mk[(size_t)idx * KD + c];
      u16 h = f2bf(x);
      mkh[(size_t)idx * KD + c] = h;
      mkl[(size_t)idx * KD + c] = f2bf(x - bf2f(h));
      ss += x * x;
    }
    red[tid] = ss; __syncthreads();
    for (int o = 128; o > 0; o >>= 1) { if (tid < o) red[tid] += red[tid + o]; __syncthreads(); }
    if (tid == 0) normk[idx] = sqrtf(red[0]);
    __syncthreads();
  }
}

// ---------------------------------------------------------------------------
// Read phase A: scores for a 16-query x 2048-memory slab (WMMA into LDS),
// fused per-row partial top-10 -> candidate buffer. Avoids a 256MB tensor.
// ---------------------------------------------------------------------------
__global__ __launch_bounds__(256) void scores_part(
    const u16* __restrict__ qh, const u16* __restrict__ ql,
    const u16* __restrict__ mkh, const u16* __restrict__ mkl,
    float* __restrict__ cval, int* __restrict__ cidxr) {
  extern __shared__ float smem_sc[];
  float* ssc = smem_sc;                       // 16 x 2048 scores
  float* pv = smem_sc + 16 * 2048;            // 16 x 32 x 10 lane candidates
  int* pi = (int*)(pv + 16 * 32 * TKR);
  const int lane = threadIdx.x & 31, w = threadIdx.x >> 5;
  const int qrow0 = blockIdx.y * 16;
  const int col0 = blockIdx.x * 2048;
  const float sc = 0.044194173824159216f;     // 1/sqrt(512)
  const int half = lane >> 4, sub = lane & 15;
  for (int g = 0; g < 4; ++g) {
    v8f acc[4] = {};
    wmma_bf16x3_multi<4>(qh, ql, KD, qrow0, mkh, mkl, KD, col0 + w * 16 + g * 512, 128, KD, lane, acc);
#pragma unroll
    for (int u = 0; u < 4; ++u) {
      const int lc = w * 16 + g * 512 + u * 128 + sub;
#pragma unroll
      for (int i = 0; i < 8; ++i) ssc[(i + half * 8) * 2048 + lc] = acc[u][i] * sc;
    }
  }
  __syncthreads();
#pragma unroll 1
  for (int rr = 0; rr < 2; ++rr) {
    const int r = w * 2 + rr;
    float tv[TKR]; int ti[TKR];
#pragma unroll
    for (int s = 0; s < TKR; ++s) { tv[s] = -3.4e38f; ti[s] = 0x7fffffff; }
    for (int c = lane; c < 2048; c += 32) tk_ins<TKR>(tv, ti, ssc[r * 2048 + c], col0 + c);
#pragma unroll
    for (int s = 0; s < TKR; ++s) { pv[(r * 32 + lane) * TKR + s] = tv[s]; pi[(r * 32 + lane) * TKR + s] = ti[s]; }
  }
  __syncthreads();
  if (lane == 0) {
#pragma unroll 1
    for (int rr = 0; rr < 2; ++rr) {
      const int r = w * 2 + rr;
      float tv[TKR]; int ti[TKR];
#pragma unroll
      for (int s = 0; s < TKR; ++s) { tv[s] = -3.4e38f; ti[s] = 0x7fffffff; }
      for (int c = 0; c < 32 * TKR; ++c) tk_ins<TKR>(tv, ti, pv[r * 32 * TKR + c], pi[r * 32 * TKR + c]);
      const size_t base = ((size_t)(qrow0 + r) * 32 + blockIdx.x) * TKR;
#pragma unroll
      for (int s = 0; s < TKR; ++s) { cval[base + s] = tv[s]; cidxr[base + s] = ti[s]; }
    }
  }
}

// ---------------------------------------------------------------------------
// Read phase B: merge 32x10 candidates -> top-10, softmax, gather mem_values
// ---------------------------------------------------------------------------
__global__ __launch_bounds__(256) void read_out(
    const float* __restrict__ cval, const int* __restrict__ cidxr,
    const float* __restrict__ mv, float* __restrict__ vals) {
  __shared__ float w10[TKR];
  __shared__ int i10[TKR];
  const int q = blockIdx.x;
  if (threadIdx.x == 0) {
    float tv[TKR]; int ti[TKR];
#pragma unroll
    for (int s = 0; s < TKR; ++s) { tv[s] = -3.4e38f; ti[s] = 0x7fffffff; }
    const size_t base = (size_t)q * 32 * TKR;
    for (int c = 0; c < 32 * TKR; ++c) tk_ins<TKR>(tv, ti, cval[base + c], cidxr[base + c]);
    float m = tv[0], ssum = 0.f, e[TKR];
#pragma unroll
    for (int s = 0; s < TKR; ++s) { e[s] = __expf(tv[s] - m); ssum += e[s]; }
#pragma unroll
    for (int s = 0; s < TKR; ++s) { w10[s] = e[s] / ssum; i10[s] = ti[s]; }
  }
  __syncthreads();
  for (int v = threadIdx.x; v < VD; v += 256) {
    float s = 0.f;
#pragma unroll
    for (int kk = 0; kk < TKR; ++kk) s += w10[kk] * mv[(size_t)i10[kk] * VD + v];
    vals[(size_t)q * VD + v] = s;
  }
}

// ---------------------------------------------------------------------------
// Device-scope barrier (sense via generation counter) for persistent LSTM
// ---------------------------------------------------------------------------
__device__ void gbar(int* cnt, int* gen, int nb) {
  __threadfence();
  __syncthreads();
  if (threadIdx.x == 0) {
    int g = __hip_atomic_load(gen, __ATOMIC_ACQUIRE, __HIP_MEMORY_SCOPE_AGENT);
    int v = __hip_atomic_fetch_add(cnt, 1, __ATOMIC_ACQ_REL, __HIP_MEMORY_SCOPE_AGENT);
    if (v == nb - 1) {
      __hip_atomic_store(cnt, 0, __ATOMIC_RELAXED, __HIP_MEMORY_SCOPE_AGENT);
      __hip_atomic_fetch_add(gen, 1, __ATOMIC_ACQ_REL, __HIP_MEMORY_SCOPE_AGENT);
    } else {
      while (__hip_atomic_load(gen, __ATOMIC_ACQUIRE, __HIP_MEMORY_SCOPE_AGENT) == g)
        __builtin_amdgcn_s_sleep(8);
    }
  }
  __syncthreads();
}

// Per-lane 16B async copy global -> LDS (CDNA5 GLOBAL_LOAD_ASYNC_TO_LDS_B128)
__device__ __forceinline__ void async_b128_to_lds(float* dst_lds, const float* src) {
  unsigned lds = (unsigned)(size_t)dst_lds;  // low 32 bits of generic addr = LDS offset
  asm volatile("global_load_async_to_lds_b128 %0, %1, off" :: "v"(lds), "v"(src) : "memory");
}
__device__ __forceinline__ void wait_asynccnt0() {
  asm volatile("s_wait_asynccnt 0x0" ::: "memory");
}

// ---------------------------------------------------------------------------
// Persistent LSTM recurrence: 16 blocks, each owns 32 hidden units and keeps
// its 128 Whh rows resident in LDS (~258KB of the 320KB/WGP CDNA5 LDS).
// Whh staged by the Tensor Data Mover (pad 2 dwords / 256 -> 516-float pitch,
// rotating LDS banks). h exchanged through L2 with a double buffer + async
// LDS loads; one device barrier per step. c stays block-local.
// Padded row layout: elem(r,k) at r*516 + k + (k>=256 ? 2 : 0).
// ---------------------------------------------------------------------------
#define WPAD 516
__global__ __launch_bounds__(256) void lstm_seq(
    const float* __restrict__ xpre, const float* __restrict__ Whh,
    float* __restrict__ Hout, float* __restrict__ hx, int* __restrict__ bar) {
  extern __shared__ float smem_ls[];
  float* whh_s = smem_ls;                 // 128 * WPAD
  float* h_s = whh_s + 128 * WPAD;        // 2 * 512
  float* g_s = h_s + 1024;                // 2 * 4 * 32
  float* c_s = g_s + 256;                 // 64
  const int tid = threadIdx.x;
  const int jb = blockIdx.x * 32;
  const int b = tid >> 7, gate = (tid >> 5) & 3, jj = tid & 31;
  const int r = gate * 32 + jj;

#if __has_builtin(__builtin_amdgcn_tensor_load_to_lds)
  if (tid < 32) {   // one wave issues 4 TDM descriptors (EXEC-independent)
#pragma unroll 1
    for (int gg = 0; gg < 4; ++gg) {
      const unsigned lds_base = (unsigned)(size_t)(whh_s + gg * 32 * WPAD);
      const unsigned long long ga =
          (unsigned long long)(size_t)(Whh + ((size_t)(gg * 512 + jb)) * 512);
      v4u g0;
      g0[0] = 1u;                                    // count=1
      g0[1] = lds_base;                              // lds_addr (bytes)
      g0[2] = (unsigned)(ga & 0xFFFFFFFFull);        // global_addr[31:0]
      g0[3] = (unsigned)((ga >> 32) & 0x1FFFFFFull) | (2u << 30);  // addr[56:32] | type=2
      v8i g1;
      g1[0] = (int)((2u << 16) | (1u << 20) | (7u << 22) | (1u << 25)); // 4B, pad 2dw/256dw
      g1[1] = (int)((16384u & 0xFFFFu) << 16);       // tensor_dim0 = 32*512
      g1[2] = (int)((16384u >> 16) | (1u << 16));    // tensor_dim0 hi | tensor_dim1 = 1
      g1[3] = (int)(16384u << 16);                   // tile_dim0 = 16384
      g1[4] = 0;                                     // tile_dim1/2 unused (1D)
      g1[5] = 16384;                                 // tensor_dim0_stride lo
      g1[6] = 0;
      g1[7] = 0;
      v4i g2 = {0, 0, 0, 0};
      v4i g3 = {0, 0, 0, 0};
      v8i g4 = {0, 0, 0, 0, 0, 0, 0, 0};
      __builtin_amdgcn_tensor_load_to_lds(g0, g1, g2, g3, g4, 0);
    }
    __builtin_amdgcn_s_wait_tensorcnt((short)0);
  }
#else
  for (int idx = tid; idx < 128 * 512; idx += 256) {
    const int rr = idx >> 9, k = idx & 511;
    const int g2 = rr >> 5, j2 = rr & 31;
    whh_s[rr * WPAD + k + (k >= 256 ? 2 : 0)] = Whh[((size_t)(g2 * 512 + jb + j2)) * 512 + k];
  }
#endif
  for (int idx = tid; idx < 1024; idx += 256) h_s[idx] = 0.f;
  if (tid < 64) c_s[tid] = 0.f;
  __syncthreads();

  const float* wr = whh_s + r * WPAD;
  const float2* w2a = (const float2*)wr;          // K = 0..255
  const float2* w2b = (const float2*)(wr + 258);  // K = 256..511 (past mid-row pad)
  const float2* h2 = (const float2*)(h_s + b * 512);
  const int nb = (int)gridDim.x;
  for (int t = 0; t < 512; ++t) {
    float* hxw = hx + (t & 1) * 1024;
    float acc = xpre[((size_t)(b * 512 + t)) * G4 + gate * 512 + jb + jj];
#pragma unroll 8
    for (int k = 0; k < 128; ++k) { float2 hv = h2[k]; float2 wv = w2a[k]; acc += hv.x * wv.x + hv.y * wv.y; }
#pragma unroll 8
    for (int k = 0; k < 128; ++k) { float2 hv = h2[128 + k]; float2 wv = w2b[k]; acc += hv.x * wv.x + hv.y * wv.y; }
    g_s[(b * 4 + gate) * 32 + jj] = acc;
    __syncthreads();
    if (tid < 64) {
      const int bb = tid >> 5, j3 = tid & 31;
      const float gi = g_s[(bb * 4 + 0) * 32 + j3];
      const float gf = g_s[(bb * 4 + 1) * 32 + j3];
      const float gg = g_s[(bb * 4 + 2) * 32 + j3];
      const float go = g_s[(bb * 4 + 3) * 32 + j3];
      const float c = sigmf(gf) * c_s[tid] + sigmf(gi) * tanhf(gg);
      const float h = sigmf(go) * tanhf(c);
      c_s[tid] = c;
      hxw[bb * 512 + jb + j3] = h;
      Hout[((size_t)(bb * 512 + t)) * 512 + jb + j3] = h;
    }
    gbar(bar, bar + 1, nb);
    // async broadcast reload of h (4KB): 256 lanes x 16B
    async_b128_to_lds(h_s + tid * 4, hxw + tid * 4);
    wait_asynccnt0();
    __syncthreads();
  }
}

// ---------------------------------------------------------------------------
// Host-side orchestration
// ---------------------------------------------------------------------------
extern "C" void kernel_launch(void* const* d_in, const int* in_sizes, int n_in,
                              void* d_out, int out_size, void* d_ws, size_t ws_size,
                              hipStream_t stream) {
  (void)in_sizes; (void)n_in; (void)out_size; (void)ws_size;
  const float* write_keys = (const float*)d_in[0];
  const float* write_values = (const float*)d_in[1];
  const float* queries = (const float*)d_in[2];
  const float* mem_keys = (const float*)d_in[3];
  const float* mem_values = (const float*)d_in[4];
  const float* Wq = (const float*)d_in[5];
  const float* Wk = (const float*)d_in[6];
  const float* Wv = (const float*)d_in[7];
  const float* lr = (const float*)d_in[8];
  const float* Wih0 = (const float*)d_in[9];
  const float* Whh0 = (const float*)d_in[10];
  const float* bih0 = (const float*)d_in[11];
  const float* bhh0 = (const float*)d_in[12];
  const float* Wih1 = (const float*)d_in[13];
  const float* Whh1 = (const float*)d_in[14];
  const float* bih1 = (const float*)d_in[15];
  const float* bhh1 = (const float*)d_in[16];
  float* out = (float*)d_out;

  char* p = (char*)d_ws;
  auto take = [&](size_t bytes) -> void* {
    void* r = (void*)p;
    p += (bytes + 255) & ~(size_t)255;
    return r;
  };
  float* mk_work = (float*)take((size_t)MMEM * KD * 4);   // mutable key bank (f32, exact)
  float* mv_work = (float*)take((size_t)MMEM * VD * 4);   // mutable value bank
  u16* mkh = (u16*)take((size_t)MMEM * KD * 2);           // bf16-hi mirror of keys
  u16* mkl = (u16*)take((size_t)MMEM * KD * 2);           // bf16-lo mirror of keys
  float* normk = (float*)take((size_t)MMEM * 4);          // per-row key norms
  float* kp = (float*)take((size_t)NKROWS * KD * 4);
  float* vp = (float*)take((size_t)NKROWS * VD * 4);
  float* qb = (float*)take((size_t)NQ * KD * 4);
  u16* qh = (u16*)take((size_t)NQ * KD * 2);
  u16* ql = (u16*)take((size_t)NQ * KD * 2);
  float* qn = (float*)take((size_t)NQ * 4);
  u16* kchh = (u16*)take((size_t)16 * KD * 2);
  u16* kchl = (u16*)take((size_t)16 * KD * 2);
  float* kn = (float*)take(64);
  float* simb = (float*)take((size_t)CHUNKSZ * MMEM * 4);
  int* cidx = (int*)take(CHUNKSZ * TKW * 4);
  float* calpha = (float*)take(CHUNKSZ * TKW * 4);
  float* cval = (float*)take((size_t)NQ * 32 * TKR * 4);
  int* cidxr = (int*)take((size_t)NQ * 32 * TKR * 4);
  float* vals = (float*)take((size_t)NQ * VD * 4);
  float* xpre = (float*)take((size_t)NQ * G4 * 4);
  float* hseq = (float*)take((size_t)NQ * VD * 4);
  float* hx = (float*)take(2 * 1024 * 4);
  int* bar = (int*)take(256);

  (void)hipMemcpyAsync(mk_work, mem_keys, (size_t)MMEM * KD * 4, hipMemcpyDeviceToDevice, stream);
  (void)hipMemcpyAsync(mv_work, mem_values, (size_t)MMEM * VD * 4, hipMemcpyDeviceToDevice, stream);
  (void)hipMemsetAsync(bar, 0, 8, stream);

  // key-bank mirrors + norms
  prep_rows<<<MMEM / 8, 256, 0, stream>>>(mk_work, mkh, mkl, normk, KD);

  // input projections (exact f32 WMMA)
  gemm_f32_wmma<<<dim3(KD / 16, NKROWS / 16), 32, 0, stream>>>(write_keys, KD, Wk, KD, kp, KD, KD, 1.f, nullptr, nullptr);
  gemm_f32_wmma<<<dim3(VD / 16, NKROWS / 16), 32, 0, stream>>>(write_values, VD, Wv, VD, vp, VD, VD, 1.f, nullptr, nullptr);
  gemm_f32_wmma<<<dim3(KD / 16, NQ / 16), 32, 0, stream>>>(queries, KD, Wq, KD, qb, KD, KD, 1.f, nullptr, nullptr);
  prep_rows<<<NQ / 8, 256, 0, stream>>>(qb, qh, ql, qn, KD);

  // sequential Hebbian write: sim(WMMA) -> top5 -> 40 sequential blends
  for (int chunk = 0; chunk < NCHUNK; ++chunk) {
    chunk_prep<<<1, 256, 0, stream>>>(kp, chunk, kchh, kchl, kn);
    hebb_sim<<<MMEM / 2048, 256, 0, stream>>>(kchh, kchl, kn, mkh, mkl, normk, simb);
    hebb_top5<<<CHUNKSZ, 256, 0, stream>>>(simb, lr, cidx, calpha);
    hebb_update<<<1, 256, 0, stream>>>(mk_work, mv_work, mkh, mkl, normk, kp, vp, cidx, calpha, chunk);
  }

  // read: fused scores(WMMA)+top10 slabs, then merge + softmax + gather
  const size_t sc_smem = (size_t)(16 * 2048 + 2 * 16 * 32 * TKR) * 4;
  scores_part<<<dim3(MMEM / 2048, NQ / 16), 256, sc_smem, stream>>>(qh, ql, mkh, mkl, cval, cidxr);
  read_out<<<NQ, 256, 0, stream>>>(cval, cidxr, mv_work, vals);

  // two-layer LSTM: hoisted input GEMMs (WMMA) + persistent recurrence
  const size_t ls_smem = (size_t)(128 * WPAD + 1024 + 256 + 64) * 4;
  gemm_f32_wmma<<<dim3(G4 / 16, NQ / 16), 32, 0, stream>>>(vals, VD, Wih0, VD, xpre, G4, VD, 1.f, bih0, bhh0);
  lstm_seq<<<16, 256, ls_smem, stream>>>(xpre, Whh0, hseq, hx, bar);
  gemm_f32_wmma<<<dim3(G4 / 16, NQ / 16), 32, 0, stream>>>(hseq, VD, Wih1, VD, xpre, G4, VD, 1.f, bih1, bhh1);
  lstm_seq<<<16, 256, ls_smem, stream>>>(xpre, Whh1, out, hx, bar);
}
